// LatticeRNN_91268055040456
// MI455X (gfx1250) — matrix operand order repaired and checked by reading, not compiled, not run on hardware
//
#include <hip/hip_runtime.h>
#include <hip/hip_bf16.h>
#include <math.h>

typedef __bf16 bf16_t;
typedef __attribute__((ext_vector_type(16))) __bf16 v16bf;
typedef __attribute__((ext_vector_type(8)))  __bf16 v8bf;
typedef __attribute__((ext_vector_type(8)))  float  v8f;

#define GHDIM 9
#define GWDIM 9
#define NCELL 81
#define BATCH 4096
#define HD 128
#define H3 384
#define H4 512
#define TILE_B 32
#define CELL_THREADS 128
#define SA_STRIDE 392   // 384 + 8 bf16 pad (784B row, 16B aligned, bank-spread)
#define PH_STRIDE 132   // 128 + 4 f32 pad
#define PHB_STRIDE 136  // 128 + 8 bf16 pad (272B row, 16B aligned)
#define LN_EPS 1e-5f

// ---------------- WMMA helpers (CDNA5 wave32) ----------------

__device__ __forceinline__ v8f wmma_bf16(v16bf a, v16bf b, v8f c) {
  // v_wmma_f32_16x16x32_bf16
  return __builtin_amdgcn_wmma_f32_16x16x32_bf16(false, a, false, b, (short)0, c,
                                                 false, false);
}

// A fragment (16x32 bf16) from LDS, row-major with leading dim `ld` (bf16 elems).
// lanes 0-15: M = m0+lane, elems 0..15 = K k0..k0+15
// lanes 16-31: same M, K shifted by +16.
__device__ __forceinline__ v16bf load_a_lds(const bf16_t* base, int ld, int m0,
                                            int k0, int lane) {
  const bf16_t* p = base + (m0 + (lane & 15)) * ld + k0 + ((lane >> 4) << 4);
  v8bf lo = *(const v8bf*)(p);
  v8bf hi = *(const v8bf*)(p + 8);
  return __builtin_shufflevector(lo, hi, 0, 1, 2, 3, 4, 5, 6, 7, 8, 9, 10, 11,
                                 12, 13, 14, 15);
}

// B fragment (32x16 bf16) from global weights stored [N][K] row-major.
// lanes 0-15: N = n0+lane, K = k0..k0+15 ; lanes 16-31: N = n0+lane-16, K = k0+16..k0+31
__device__ __forceinline__ v16bf load_b_glb(const bf16_t* w, int ldk, int n0,
                                            int k0, int lane) {
  const bf16_t* p = w + (size_t)(n0 + (lane & 15)) * ldk + k0 + ((lane >> 4) << 4);
  return *(const v16bf*)p;
}

__device__ __forceinline__ float sigf(float v) { return 1.f / (1.f + expf(-v)); }

// ---------------- weight pre-pack f32 -> bf16 ----------------

__global__ void pack_weights(const float* __restrict__ Wh,
                             const float* __restrict__ Wc,
                             const float* __restrict__ Whh,
                             bf16_t* __restrict__ whb, bf16_t* __restrict__ wcb,
                             bf16_t* __restrict__ whhb) {
  const size_t n1 = (size_t)NCELL * HD * H3;   // Wh / Wc
  const size_t n3 = (size_t)NCELL * H4 * HD;   // W_hh
  const size_t total = n1 + n1 + n3;
  for (size_t idx = (size_t)blockIdx.x * blockDim.x + threadIdx.x; idx < total;
       idx += (size_t)gridDim.x * blockDim.x) {
    if (idx < n1)
      whb[idx] = (bf16_t)Wh[idx];
    else if (idx < 2 * n1)
      wcb[idx - n1] = (bf16_t)Wc[idx - n1];
    else
      whhb[idx - 2 * n1] = (bf16_t)Whh[idx - 2 * n1];
  }
}

// ---------------- wavefront cell kernel ----------------

struct CellArgs {
  const float* x;
  const float* h_ext;
  const float* c_ext;
  const float* h_grid0;
  const float* c_grid0;
  const float* bh;
  const float* bc;
  const float* lnh_g;
  const float* lnh_b;
  const float* lnc_g;
  const float* lnc_b;
  const float* w_ih;
  const float* b_ih;
  const float* b_hh;
  const bf16_t* whb;
  const bf16_t* wcb;
  const bf16_t* whhb;
  float* fr_h;
  float* fr_c;
  int step;
};

__global__ __launch_bounds__(CELL_THREADS) void cell_kernel(CellArgs a) {
  __shared__ __align__(32) unsigned char smem_raw[TILE_B * SA_STRIDE * 2 +
                                                  TILE_B * PH_STRIDE * 4 * 2];
  bf16_t* sA = (bf16_t*)smem_raw;    // 25088 B (staged comb, bf16)
  bf16_t* sPHB = (bf16_t*)smem_raw;  // overlay: LN'd ph in bf16 (8704 B)
  float* sPH = (float*)(smem_raw + TILE_B * SA_STRIDE * 2);
  float* sPC = (float*)(smem_raw + TILE_B * SA_STRIDE * 2 + TILE_B * PH_STRIDE * 4);

  const int s = a.step;
  const int i0 = (s > 8) ? (s - 8) : 0;
  const int gi = i0 + blockIdx.y;
  const int gj = s - gi;
  const int k = gi * GWDIM + gj;
  const int b0 = blockIdx.x * TILE_B;
  const int tid = threadIdx.x;
  const int lane = tid & 31;
  const int wid = tid >> 5;  // 0..3 : column group (32 cols each for GEMM1/2)
  const int rp = (s + 1) & 1;  // read parity (step s-1 wrote here)
  const int wp = s & 1;        // write parity
  const size_t plane = (size_t)BATCH * HD;

  const float* hl = (gi == 0 && gj == 0)
                        ? a.h_ext
                        : ((gj > 0) ? a.fr_h + ((size_t)rp * GWDIM + (gj - 1)) * plane
                                    : nullptr);
  const float* hu = (gi > 0) ? a.fr_h + ((size_t)rp * GWDIM + gj) * plane : nullptr;
  const float* hp = a.h_grid0 + (size_t)k * plane;
  const float* cl = (gi == 0 && gj == 0)
                        ? a.c_ext
                        : ((gj > 0) ? a.fr_c + ((size_t)rp * GWDIM + (gj - 1)) * plane
                                    : nullptr);
  const float* cu = (gi > 0) ? a.fr_c + ((size_t)rp * GWDIM + gj) * plane : nullptr;
  const float* cp = a.c_grid0 + (size_t)k * plane;

  const bf16_t* whb_k = a.whb + (size_t)k * HD * H3;
  const bf16_t* wcb_k = a.wcb + (size_t)k * HD * H3;
  const bf16_t* whh_k = a.whhb + (size_t)k * H4 * HD;

  // ---- two passes: 0 = h path -> sPH, 1 = c path -> sPC ----
  for (int pass = 0; pass < 2; ++pass) {
    const float* s0 = pass ? cl : hl;
    const float* s1 = pass ? cu : hu;
    const float* s2 = pass ? cp : hp;
    const bf16_t* wb = pass ? wcb_k : whb_k;
    const float* bias = (pass ? a.bc : a.bh) + k * HD;
    float* sOut = pass ? sPC : sPH;

    // stage comb (bf16) into LDS: [TILE_B][384]
    for (int idx = tid; idx < TILE_B * H3; idx += CELL_THREADS) {
      int r = idx / H3;
      int c = idx - r * H3;
      const float* src = (c < HD) ? s0 : ((c < 2 * HD) ? s1 : s2);
      float v = src ? src[(size_t)(b0 + r) * HD + (c & (HD - 1))] : 0.f;
      sA[r * SA_STRIDE + c] = (bf16_t)v;
    }
    __syncthreads();

    // 2x2 register tile per wave: rows {0..15,16..31}, cols wid*32 + {0..15,16..31}
    v8f acc[2][2];
    for (int mt = 0; mt < 2; ++mt)
      for (int nt = 0; nt < 2; ++nt) acc[mt][nt] = (v8f){0, 0, 0, 0, 0, 0, 0, 0};
    for (int k0 = 0; k0 < H3; k0 += 32) {
      v16bf af0 = load_a_lds(sA, SA_STRIDE, 0, k0, lane);
      v16bf af1 = load_a_lds(sA, SA_STRIDE, 16, k0, lane);
      v16bf bf0 = load_b_glb(wb, H3, wid * 32, k0, lane);
      v16bf bf1 = load_b_glb(wb, H3, wid * 32 + 16, k0, lane);
      acc[0][0] = wmma_bf16(af0, bf0, acc[0][0]);
      acc[0][1] = wmma_bf16(af0, bf1, acc[0][1]);
      acc[1][0] = wmma_bf16(af1, bf0, acc[1][0]);
      acc[1][1] = wmma_bf16(af1, bf1, acc[1][1]);
    }
    // write accumulators + bias to LDS f32 (C-layout: M = v + (lane>=16)*8, N = lane&15)
    for (int mt = 0; mt < 2; ++mt)
      for (int nt = 0; nt < 2; ++nt) {
        int n = wid * 32 + nt * 16 + (lane & 15);
        float bb = bias[n];
        for (int v = 0; v < 8; ++v) {
          int m = mt * 16 + v + ((lane >> 4) << 3);
          sOut[m * PH_STRIDE + n] = acc[mt][nt][v] + bb;
        }
      }
    __syncthreads();  // also guards sA restage for next pass
  }

  // ---- LayerNorm: ph -> sPHB (bf16, overlays sA); pc normalized in place ----
  if (tid < TILE_B) {
    const float* gh = a.lnh_g + k * HD;
    const float* bhv = a.lnh_b + k * HD;
    const float* gc = a.lnc_g + k * HD;
    const float* bcv = a.lnc_b + k * HD;
    float* ph = sPH + tid * PH_STRIDE;
    float* pc = sPC + tid * PH_STRIDE;
    bf16_t* pb = sPHB + tid * PHB_STRIDE;

    float mu = 0.f;
    for (int c = 0; c < HD; ++c) mu += ph[c];
    mu *= (1.f / HD);
    float va = 0.f;
    for (int c = 0; c < HD; ++c) {
      float d = ph[c] - mu;
      va += d * d;
    }
    float inv = rsqrtf(va * (1.f / HD) + LN_EPS);
    for (int c = 0; c < HD; ++c)
      pb[c] = (bf16_t)((ph[c] - mu) * inv * gh[c] + bhv[c]);

    mu = 0.f;
    for (int c = 0; c < HD; ++c) mu += pc[c];
    mu *= (1.f / HD);
    va = 0.f;
    for (int c = 0; c < HD; ++c) {
      float d = pc[c] - mu;
      va += d * d;
    }
    inv = rsqrtf(va * (1.f / HD) + LN_EPS);
    for (int c = 0; c < HD; ++c) pc[c] = (pc[c] - mu) * inv * gc[c] + bcv[c];
  }
  __syncthreads();

  // ---- gate GEMM: gates = ph @ W_hh^T (+ x*W_ih + b_ih + b_hh), then LSTM gating ----
  const float* wih = a.w_ih + (size_t)k * H4;
  const float* bih = a.b_ih + (size_t)k * H4;
  const float* bhh2 = a.b_hh + (size_t)k * H4;
  float* frh_w = a.fr_h + ((size_t)wp * GWDIM + gj) * plane;
  float* frc_w = a.fr_c + ((size_t)wp * GWDIM + gj) * plane;

  for (int ht = 0; ht < 2; ++ht) {
    int hc0 = wid * 32 + ht * 16;  // h-column tile for this wave
    // 2 M-strips x 4 gates, B fragment shared across M-strips
    v8f g[2][4];
    for (int ms = 0; ms < 2; ++ms)
      for (int gg = 0; gg < 4; ++gg) g[ms][gg] = (v8f){0, 0, 0, 0, 0, 0, 0, 0};
    for (int k0 = 0; k0 < HD; k0 += 32) {
      v16bf a0 = load_a_lds(sPHB, PHB_STRIDE, 0, k0, lane);
      v16bf a1 = load_a_lds(sPHB, PHB_STRIDE, 16, k0, lane);
      for (int gg = 0; gg < 4; ++gg) {
        v16bf bf = load_b_glb(whh_k, HD, gg * HD + hc0, k0, lane);
        g[0][gg] = wmma_bf16(a0, bf, g[0][gg]);
        g[1][gg] = wmma_bf16(a1, bf, g[1][gg]);
      }
    }
    for (int ms = 0; ms < 2; ++ms) {
      for (int v = 0; v < 8; ++v) {
        int m = ms * 16 + v + ((lane >> 4) << 3);
        int n = hc0 + (lane & 15);
        size_t b = (size_t)(b0 + m);
        float xv = a.x[b * (GHDIM * GWDIM) + k];
        float ig = g[ms][0][v] + xv * wih[0 * HD + n] + bih[0 * HD + n] + bhh2[0 * HD + n];
        float fg = g[ms][1][v] + xv * wih[1 * HD + n] + bih[1 * HD + n] + bhh2[1 * HD + n];
        float gv = g[ms][2][v] + xv * wih[2 * HD + n] + bih[2 * HD + n] + bhh2[2 * HD + n];
        float og = g[ms][3][v] + xv * wih[3 * HD + n] + bih[3 * HD + n] + bhh2[3 * HD + n];
        float pcv = sPC[m * PH_STRIDE + n];
        float cn = sigf(fg) * pcv + sigf(ig) * tanhf(gv);
        float hn = sigf(og) * tanhf(cn);
        frh_w[b * HD + n] = hn;
        frc_w[b * HD + n] = cn;
      }
    }
  }
}

// ---------------- head ----------------

__global__ void prep_z0(const float* __restrict__ frh, const float* __restrict__ frc,
                        const float* __restrict__ h_ext,
                        const float* __restrict__ c_ext, float* __restrict__ z0) {
  int idx = blockIdx.x * 256 + threadIdx.x;  // 4096*256 threads
  int r = idx >> 8;
  int c = idx & 255;
  float v;
  if (c < 128)
    v = frh[(size_t)r * 128 + c] + h_ext[(size_t)r * 128 + c];
  else
    v = frc[(size_t)r * 128 + (c - 128)] + c_ext[(size_t)r * 128 + (c - 128)];
  z0[idx] = v;
}

// out[b,o] = sum_n in[b,n]*W[o,n] + bias[o]; blockDim.x == OUTF; 16 rows/block
__global__ void head_fc(const float* __restrict__ in, const float* __restrict__ W,
                        const float* __restrict__ bias, float* __restrict__ out,
                        int INF, int OUTF) {
  __shared__ float zt[16 * 256];
  int b0 = blockIdx.x * 16;
  int tid = threadIdx.x;
  int nth = blockDim.x;
  for (int idx = tid; idx < 16 * INF; idx += nth) {
    int r = idx / INF;
    int c = idx - r * INF;
    zt[r * INF + c] = in[(size_t)(b0 + r) * INF + c];
  }
  __syncthreads();
  int o = tid;
  float acc[16];
  float bb = bias[o];
  for (int r = 0; r < 16; ++r) acc[r] = bb;
  for (int n = 0; n < INF; ++n) {
    float w = W[(size_t)o * INF + n];
    for (int r = 0; r < 16; ++r) acc[r] += zt[r * INF + n] * w;
  }
  for (int r = 0; r < 16; ++r) out[(size_t)(b0 + r) * OUTF + o] = acc[r];
}

__global__ void head_fc3(const float* __restrict__ in, const float* __restrict__ W,
                         const float* __restrict__ bias, float* __restrict__ out) {
  int b = blockIdx.x * 256 + threadIdx.x;
  if (b < BATCH) {
    float acc = bias[0];
    for (int n = 0; n < 128; ++n) acc += in[(size_t)b * 128 + n] * W[n];
    out[b] = acc;
  }
}

// BatchNorm1d (training stats over batch) + activation. One feature per block.
__global__ void bn_act(const float* __restrict__ in, const float* __restrict__ g,
                       const float* __restrict__ b, float* __restrict__ out, int B,
                       int F, int act) {
  const int f = blockIdx.x;
  const int tid = threadIdx.x;
  __shared__ float red[256];
  __shared__ float mu_s, inv_s;
  float s = 0.f;
  for (int ii = tid; ii < B; ii += 256) s += in[(size_t)ii * F + f];
  red[tid] = s;
  __syncthreads();
  for (int st = 128; st > 0; st >>= 1) {
    if (tid < st) red[tid] += red[tid + st];
    __syncthreads();
  }
  if (tid == 0) mu_s = red[0] / (float)B;
  __syncthreads();
  float mu = mu_s;
  float s2 = 0.f;
  for (int ii = tid; ii < B; ii += 256) {
    float d = in[(size_t)ii * F + f] - mu;
    s2 += d * d;
  }
  __syncthreads();
  red[tid] = s2;
  __syncthreads();
  for (int st = 128; st > 0; st >>= 1) {
    if (tid < st) red[tid] += red[tid + st];
    __syncthreads();
  }
  if (tid == 0) inv_s = rsqrtf(red[0] / (float)B + LN_EPS);
  __syncthreads();
  float inv = inv_s, gg = g[f], bb = b[f];
  for (int ii = tid; ii < B; ii += 256) {
    float v = (in[(size_t)ii * F + f] - mu) * inv * gg + bb;
    out[(size_t)ii * F + f] = act ? (1.f / (1.f + expf(-v))) : fmaxf(v, 0.f);
  }
}

// ---------------- launch ----------------

extern "C" void kernel_launch(void* const* d_in, const int* in_sizes, int n_in,
                              void* d_out, int out_size, void* d_ws, size_t ws_size,
                              hipStream_t stream) {
  (void)in_sizes;
  (void)n_in;
  (void)out_size;
  (void)ws_size;
  const float* x = (const float*)d_in[0];
  const float* h_ext = (const float*)d_in[1];
  const float* c_ext = (const float*)d_in[2];
  const float* h_grid0 = (const float*)d_in[3];
  const float* c_grid0 = (const float*)d_in[4];
  const float* Wh = (const float*)d_in[5];
  const float* bh = (const float*)d_in[6];
  const float* Wc = (const float*)d_in[7];
  const float* bc = (const float*)d_in[8];
  const float* lnh_g = (const float*)d_in[9];
  const float* lnh_b = (const float*)d_in[10];
  const float* lnc_g = (const float*)d_in[11];
  const float* lnc_b = (const float*)d_in[12];
  const float* W_ih = (const float*)d_in[13];
  const float* b_ih = (const float*)d_in[14];
  const float* W_hh = (const float*)d_in[15];
  const float* b_hh = (const float*)d_in[16];
  const float* fc1_W = (const float*)d_in[17];
  const float* fc1_b = (const float*)d_in[18];
  const float* bn1_g = (const float*)d_in[19];
  const float* bn1_b = (const float*)d_in[20];
  const float* fc2_W = (const float*)d_in[21];
  const float* fc2_b = (const float*)d_in[22];
  const float* bn2_g = (const float*)d_in[23];
  const float* bn2_b = (const float*)d_in[24];
  const float* fc3_W = (const float*)d_in[25];
  const float* fc3_b = (const float*)d_in[26];
  const float* bno_g = (const float*)d_in[27];
  const float* bno_b = (const float*)d_in[28];

  char* ws = (char*)d_ws;
  size_t off = 0;
  auto walloc = [&](size_t bytes) -> void* {
    void* p = ws + off;
    off = (off + bytes + 255) & ~(size_t)255;
    return p;
  };
  bf16_t* whb = (bf16_t*)walloc((size_t)NCELL * HD * H3 * sizeof(bf16_t));
  bf16_t* wcb = (bf16_t*)walloc((size_t)NCELL * HD * H3 * sizeof(bf16_t));
  bf16_t* whhb = (bf16_t*)walloc((size_t)NCELL * H4 * HD * sizeof(bf16_t));
  float* fr_h = (float*)walloc((size_t)2 * GWDIM * BATCH * HD * sizeof(float));
  float* fr_c = (float*)walloc((size_t)2 * GWDIM * BATCH * HD * sizeof(float));
  float* z0 = (float*)walloc((size_t)BATCH * 256 * sizeof(float));
  float* zA = (float*)walloc((size_t)BATCH * 256 * sizeof(float));
  float* zB = (float*)walloc((size_t)BATCH * 128 * sizeof(float));
  float* zC = (float*)walloc((size_t)BATCH * sizeof(float));

  pack_weights<<<4096, 256, 0, stream>>>(Wh, Wc, W_hh, whb, wcb, whhb);

  CellArgs ca;
  ca.x = x;
  ca.h_ext = h_ext;
  ca.c_ext = c_ext;
  ca.h_grid0 = h_grid0;
  ca.c_grid0 = c_grid0;
  ca.bh = bh;
  ca.bc = bc;
  ca.lnh_g = lnh_g;
  ca.lnh_b = lnh_b;
  ca.lnc_g = lnc_g;
  ca.lnc_b = lnc_b;
  ca.w_ih = W_ih;
  ca.b_ih = b_ih;
  ca.b_hh = b_hh;
  ca.whb = whb;
  ca.wcb = wcb;
  ca.whhb = whhb;
  ca.fr_h = fr_h;
  ca.fr_c = fr_c;

  for (int s = 0; s < GHDIM + GWDIM - 1; ++s) {
    int i0 = (s > 8) ? (s - 8) : 0;
    int i1 = (s < 8) ? s : 8;
    int ncells = i1 - i0 + 1;
    ca.step = s;
    dim3 grid(BATCH / TILE_B, ncells);
    cell_kernel<<<grid, CELL_THREADS, 0, stream>>>(ca);
  }

  // step 16 writes parity 0, column j=8
  const float* frh_final = fr_h + ((size_t)0 * GWDIM + 8) * (size_t)BATCH * HD;
  const float* frc_final = fr_c + ((size_t)0 * GWDIM + 8) * (size_t)BATCH * HD;

  prep_z0<<<(BATCH * 256) / 256, 256, 0, stream>>>(frh_final, frc_final, h_ext,
                                                   c_ext, z0);
  head_fc<<<BATCH / 16, 256, 0, stream>>>(z0, fc1_W, fc1_b, zA, 256, 256);
  bn_act<<<256, 256, 0, stream>>>(zA, bn1_g, bn1_b, zA, BATCH, 256, 0);
  head_fc<<<BATCH / 16, 128, 0, stream>>>(zA, fc2_W, fc2_b, zB, 256, 128);
  bn_act<<<128, 256, 0, stream>>>(zB, bn2_g, bn2_b, zB, BATCH, 128, 0);
  head_fc3<<<BATCH / 256, 256, 0, stream>>>(zB, fc3_W, fc3_b, zC);
  bn_act<<<1, 256, 0, stream>>>(zC, bno_g, bno_b, (float*)d_out, BATCH, 1, 1);
}
